// NeRank_48421461295167
// MI455X (gfx1250) — matrix operand order repaired and checked by reading, not compiled, not run on hardware
//
#include <hip/hip_runtime.h>
#include <hip/hip_bf16.h>
#include <stdint.h>

// ---------------------------------------------------------------------------
// CDNA5 WMMA (wave32): D(16x16 f32) = A(16x32 bf16) x B(32x16 bf16) + C
// ---------------------------------------------------------------------------
typedef __attribute__((ext_vector_type(16))) __bf16 v16bf;
typedef __attribute__((ext_vector_type(8)))  float  v8f;
typedef unsigned int u32x4 __attribute__((ext_vector_type(4)));
typedef int          i32x8 __attribute__((ext_vector_type(8)));
typedef int          i32x4 __attribute__((ext_vector_type(4)));

#define LSEQ 64
#define IDIM 300
#define IPAD 320     // K padded to multiple of 32 for WMMA
#define HDIM 128
#define G4   512     // 4*H gate width
#define NSEQ 12288   // 2048 qu + 2048*5 qn sequences
#define NQU  2048

__device__ __forceinline__ unsigned short f2bf(float f) {
  union { float f; unsigned u; } x; x.f = f;
  unsigned u = x.u;
  unsigned r = (u + 0x7FFFu + ((u >> 16) & 1u)) >> 16;   // RNE
  return (unsigned short)r;
}

__device__ __forceinline__ float sigm(float x) { return 1.0f / (1.0f + __expf(-x)); }

__device__ __forceinline__ float tanh_fast(float x) {
  float e = __expf(-2.0f * fabsf(x));
  float r = (1.0f - e) / (1.0f + e);
  return x < 0.0f ? -r : r;
}

__device__ __forceinline__ float log_sigmoid(float x) {
  return fminf(x, 0.0f) - log1pf(__expf(-fabsf(x)));
}

// ---------------------------------------------------------------------------
// Pack a row-major f32 weight W[Kreal x N] into WMMA B-fragment layout:
// fragment(kt,nt) = 32 lanes x 16 bf16; lane l, vgpr j holds
// B[kt*32 + 2j + 16*(l>>4) + {0,1}, nt*16 + (l&15)]  (contiguous k order, same as load_a_frag)
// ---------------------------------------------------------------------------
__global__ void pack_w_bf16(const float* __restrict__ W, unsigned short* __restrict__ out,
                            int Kreal, int Kpad, int N) {
  int KT = Kpad >> 5, NT = N >> 4;
  int total = KT * NT * 32;
  int idx = blockIdx.x * blockDim.x + threadIdx.x;
  if (idx >= total) return;
  int lane = idx & 31;
  int frag = idx >> 5;
  int nt = frag % NT;
  int kt = frag / NT;
  int hi = lane >> 4;
  int col = nt * 16 + (lane & 15);
  unsigned short* o = out + (size_t)frag * 512 + lane * 16;
#pragma unroll
  for (int j = 0; j < 8; ++j) {
    int k = kt * 32 + j * 2 + hi * 16;
    float f0 = (k     < Kreal) ? W[(size_t)k       * N + col] : 0.0f;
    float f1 = (k + 1 < Kreal) ? W[(size_t)(k + 1) * N + col] : 0.0f;
    o[2 * j]     = f2bf(f0);
    o[2 * j + 1] = f2bf(f1);
  }
}

// A-fragment gather (16-bit A 16x32, contiguous k order as in pack_w_bf16)
__device__ __forceinline__ v16bf load_a_frag(const unsigned short* base, int stride,
                                             int kt, int lane) {
  int row = lane & 15, hi = lane >> 4;
  union { v16bf v; unsigned u[8]; } r;
  const unsigned short* p = base + (size_t)row * stride + kt * 32;
#pragma unroll
  for (int j = 0; j < 8; ++j) {
    int k = j * 2 + hi * 16;
    r.u[j] = *(const unsigned*)(p + k);
  }
  return r.v;
}

// Tensor Data Mover: DMA a [2 x 300] f32 tile (row stride LSEQ*IDIM elements)
// from global to LDS. D# per ISA 8.3/8.4: count=1, type=2, data_size=4B.
// amdgpu-toolchain (clang-23) 6-arg builtin form.
__device__ __forceinline__ void issue_tdm_2row(const float* gptr, unsigned lds_off) {
  unsigned long long va = (unsigned long long)(uintptr_t)gptr;
  u32x4 g0;
  g0[0] = 1u;                                               // count=1 (user D#)
  g0[1] = lds_off;                                          // lds_addr (bytes)
  g0[2] = (unsigned)(va & 0xFFFFFFFFu);                     // global_addr lo
  g0[3] = (unsigned)((va >> 32) & 0x01FFFFFFu) | 0x80000000u; // addr hi | type=2
  i32x8 g1;
  g1[0] = 0x00020000;          // workgroup_mask=0, data_size=2 (4 bytes)
  g1[1] = (int)(IDIM << 16);   // tensor_dim0[15:0] in bits 63:48
  g1[2] = (int)(2 << 16);      // tensor_dim0 hi=0 | tensor_dim1=2
  g1[3] = (int)(IDIM << 16);   // tensor_dim1 hi=0 | tile_dim0=300
  g1[4] = 2;                   // tile_dim1=2, tile_dim2=0
  g1[5] = LSEQ * IDIM;         // tensor_dim0_stride lo32 = 19200
  g1[6] = 0;                   // stride hi, dim1_stride lo
  g1[7] = 0;
  i32x4 z4 = {0, 0, 0, 0};
  i32x8 z8 = {0, 0, 0, 0, 0, 0, 0, 0};
  __builtin_amdgcn_tensor_load_to_lds(g0, g1, z4, z4, z8, 0);
}

// ---------------------------------------------------------------------------
// Fused BiLSTM: one workgroup = 16 sequences x one direction, 8 wave32s.
// Wave w owns gate N-tiles {w, w+8, w+16, w+24} (i/f/g/o columns for
// h columns [16w,16w+16)); c-state lives in f32 registers for all 64 steps.
// x_t is TDM-double-buffered into LDS (f32), converted to bf16 per wave.
// ---------------------------------------------------------------------------
__global__ __launch_bounds__(256) void lstm_kernel(
    const float* __restrict__ qu_seq, const float* __restrict__ qn_seq,
    const int* __restrict__ qu_len, const int* __restrict__ qn_len,
    const unsigned short* __restrict__ wih, const unsigned short* __restrict__ whh,
    const float* __restrict__ b_f, const float* __restrict__ b_b,
    unsigned short* __restrict__ hcat) {
  __shared__ unsigned short xs[16][IPAD];   // x_t tile, bf16, K-padded
  __shared__ unsigned short hs[16][HDIM];   // h tile, bf16
  __shared__ float xf[2][16 * IDIM];        // TDM double buffer, f32

  const int tid  = threadIdx.x;
  const int lane = tid & 31;
  const int hi   = lane >> 4;
  const int cn   = lane & 15;
  const int wid  = __builtin_amdgcn_readfirstlane((int)(threadIdx.x >> 5)); // scalar wave id

  const int dir  = blockIdx.x & 1;
  const int tile = blockIdx.x >> 1;
  const int seq0 = tile * 16;

  const float* xbase;
  const int* lens;
  if (seq0 < NQU) { xbase = qu_seq + (size_t)seq0 * LSEQ * IDIM;         lens = qu_len + seq0; }
  else            { xbase = qn_seq + (size_t)(seq0 - NQU) * LSEQ * IDIM; lens = qn_len + (seq0 - NQU); }
  // this wave DMAs + converts rows [2*wid, 2*wid+2)
  const float* xwave = xbase + (size_t)(wid * 2) * LSEQ * IDIM;

  const unsigned short* Wd = wih + (size_t)dir * (IPAD * G4);
  const unsigned short* Ud = whh + (size_t)dir * (HDIM * G4);
  const float* bias = dir ? b_b : b_f;

  // persistent bias splat: C operand of the first (peeled) WMMA per gate
  v8f bias_splat[4];
#pragma unroll
  for (int g = 0; g < 4; ++g) {
    float bv = bias[g * 128 + wid * 16 + cn];
#pragma unroll
    for (int j = 0; j < 8; ++j) bias_splat[g][j] = bv;
  }
  int lenj[8];
#pragma unroll
  for (int j = 0; j < 8; ++j) lenj[j] = lens[j + hi * 8];

  float cst[8], hst[8];
#pragma unroll
  for (int j = 0; j < 8; ++j) { cst[j] = 0.0f; hst[j] = 0.0f; }

  unsigned short* xsf = &xs[0][0];
  unsigned short* hsf = &hs[0][0];
  for (int i = tid; i < 16 * IPAD; i += 256) xsf[i] = 0;  // zero incl. K-pad
  for (int i = tid; i < 16 * HDIM; i += 256) hsf[i] = 0;  // h0 = 0

  // LDS byte offsets for this wave's 2-row slice of each x buffer
  const unsigned xf_off0 = (unsigned)(size_t)(void*)&xf[0][wid * 2 * IDIM];
  const unsigned xf_off1 = (unsigned)(size_t)(void*)&xf[1][wid * 2 * IDIM];

  // preload x_{t(0)} into buffer 0
  {
    int t0 = dir ? (LSEQ - 1) : 0;
    issue_tdm_2row(xwave + (size_t)t0 * IDIM, xf_off0);
  }

#pragma clang loop unroll(disable)
  for (int s = 0; s < LSEQ; ++s) {
    int t = dir ? (LSEQ - 1 - s) : s;
    int cur = s & 1;

    // DMA next step's rows into the other buffer, keep <=1 TDM in flight
    if (s + 1 < LSEQ) {
      int tn = dir ? (LSEQ - 2 - s) : (s + 1);
      issue_tdm_2row(xwave + (size_t)tn * IDIM, cur ? xf_off0 : xf_off1);
      __builtin_amdgcn_s_wait_tensorcnt((short)1);
    } else {
      __builtin_amdgcn_s_wait_tensorcnt((short)0);
    }

    // convert own rows f32 -> bf16 (no cross-wave dependency on xf)
    const float* xc = &xf[cur][0];
#pragma unroll
    for (int rr = 0; rr < 2; ++rr) {
      int r = wid * 2 + rr;
      for (int k = lane; k < IDIM; k += 32)
        xs[r][k] = f2bf(xc[r * IDIM + k]);
    }
    __syncthreads();  // xs (all rows) + hs (prev step) visible to all waves

    v8f acc[4];
    // peeled kt=0: bias folds into C operand, no accumulator-init movs
    {
      v16bf a = load_a_frag(xsf, IPAD, 0, lane);
#pragma unroll
      for (int g = 0; g < 4; ++g) {
        v16bf b = *(const v16bf*)(Wd + ((size_t)(g * 8 + wid) * 32 + lane) * 16);
        acc[g] = __builtin_amdgcn_wmma_f32_16x16x32_bf16(false, a, false, b,
                                                         (short)0, bias_splat[g], false, false);
      }
    }
    // gates += x_t @ Wih  (K tiles 1..9) -- no unroll: keeps VGPRs < 256, no spills
#pragma clang loop unroll(disable)
    for (int kt = 1; kt < IPAD / 32; ++kt) {
      v16bf a = load_a_frag(xsf, IPAD, kt, lane);
#pragma unroll
      for (int g = 0; g < 4; ++g) {
        v16bf b = *(const v16bf*)(Wd + ((size_t)(kt * 32 + g * 8 + wid) * 32 + lane) * 16);
        acc[g] = __builtin_amdgcn_wmma_f32_16x16x32_bf16(false, a, false, b,
                                                         (short)0, acc[g], false, false);
      }
    }
    // gates += h @ Whh  (K = 128, 4 k-tiles)
#pragma clang loop unroll(disable)
    for (int kt = 0; kt < HDIM / 32; ++kt) {
      v16bf a = load_a_frag(hsf, HDIM, kt, lane);
#pragma unroll
      for (int g = 0; g < 4; ++g) {
        v16bf b = *(const v16bf*)(Ud + ((size_t)(kt * 32 + g * 8 + wid) * 32 + lane) * 16);
        acc[g] = __builtin_amdgcn_wmma_f32_16x16x32_bf16(false, a, false, b,
                                                         (short)0, acc[g], false, false);
      }
    }
    __syncthreads();  // all waves done reading hs before overwrite

    // activations + masked state update; publish h as bf16 for next step
#pragma unroll
    for (int j = 0; j < 8; ++j) {
      float ig = sigm(acc[0][j]);
      float fg = sigm(acc[1][j]);
      float gg = tanh_fast(acc[2][j]);
      float og = sigm(acc[3][j]);
      float c_new = fg * cst[j] + ig * gg;
      float h_new = og * tanh_fast(c_new);
      if (t < lenj[j]) { cst[j] = c_new; hst[j] = h_new; }
      hs[j + hi * 8][wid * 16 + cn] = f2bf(hst[j]);
    }
  }

  // hcat[seq][0:128]=h_f, [128:256]=h_b (bf16, FC input)
#pragma unroll
  for (int j = 0; j < 8; ++j) {
    int r = j + hi * 8;
    hcat[((size_t)(seq0 + r)) * 256 + dir * 128 + wid * 16 + cn] = f2bf(hst[j]);
  }
}

// ---------------------------------------------------------------------------
// FC: emb[12288,128] = hcat[12288,256] @ fc_W + fc_b  (WMMA, K=256)
// ---------------------------------------------------------------------------
__global__ __launch_bounds__(256) void fc_kernel(
    const unsigned short* __restrict__ hcat, const unsigned short* __restrict__ fcw,
    const float* __restrict__ fcb, float* __restrict__ emb) {
  const int tile = blockIdx.x;          // 16 rows per tile
  const int wave = threadIdx.x >> 5;    // ntile (8 x 16 = 128 cols)
  const int lane = threadIdx.x & 31;
  const int hi = lane >> 4, cn = lane & 15;

  const unsigned short* abase = hcat + (size_t)tile * 16 * 256;
  v8f bsp;
  float bv = fcb[wave * 16 + cn];
#pragma unroll
  for (int j = 0; j < 8; ++j) bsp[j] = bv;

  v8f acc;
  {
    v16bf a = load_a_frag(abase, 256, 0, lane);
    v16bf b = *(const v16bf*)(fcw + ((size_t)wave * 32 + lane) * 16);
    acc = __builtin_amdgcn_wmma_f32_16x16x32_bf16(false, a, false, b, (short)0, bsp, false, false);
  }
#pragma clang loop unroll(disable)
  for (int kt = 1; kt < 8; ++kt) {
    v16bf a = load_a_frag(abase, 256, kt, lane);
    v16bf b = *(const v16bf*)(fcw + ((size_t)(kt * 8 + wave) * 32 + lane) * 16);
    acc = __builtin_amdgcn_wmma_f32_16x16x32_bf16(false, a, false, b, (short)0, acc, false, false);
  }
#pragma unroll
  for (int j = 0; j < 8; ++j)
    emb[((size_t)(tile * 16 + j + hi * 8)) * 128 + wave * 16 + cn] = acc[j];
}

// ---------------------------------------------------------------------------
// Per-batch-row partial scores (deterministic block reduction, no atomics)
// ---------------------------------------------------------------------------
__global__ __launch_bounds__(128) void score_kernel(
    const int* __restrict__ ru_idx, const int* __restrict__ au_idx,
    const int* __restrict__ rv_idx, const int* __restrict__ av_idx,
    const int* __restrict__ rn_idx, const int* __restrict__ an_idx,
    const float* __restrict__ ru_emb, const float* __restrict__ au_emb,
    const float* __restrict__ rv_emb, const float* __restrict__ av_emb,
    const float* __restrict__ emb,
    float* __restrict__ pos_part, float* __restrict__ neg_part) {
  const int b = blockIdx.x;
  const int d = threadIdx.x;  // 0..127

  float equ = emb[(size_t)b * 128 + d];
  float eu = ru_emb[(size_t)ru_idx[b] * 128 + d] + au_emb[(size_t)au_idx[b] * 128 + d] + equ;
  float ev = rv_emb[(size_t)rv_idx[b] * 128 + d] + av_emb[(size_t)av_idx[b] * 128 + d] + equ;

  float pos = eu * ev;
  float neg = 0.0f;
#pragma unroll
  for (int k = 0; k < 5; ++k) {
    int idx = b * 5 + k;
    float nv = rv_emb[(size_t)rn_idx[idx] * 128 + d] + av_emb[(size_t)an_idx[idx] * 128 + d] +
               emb[(size_t)(NQU + idx) * 128 + d];
    neg += nv * eu;
  }

  __shared__ float sp[128], sn[128];
  sp[d] = pos; sn[d] = neg;
  __syncthreads();
  for (int off = 64; off >= 1; off >>= 1) {
    if (d < off) { sp[d] += sp[d + off]; sn[d] += sn[d + off]; }
    __syncthreads();
  }
  if (d == 0) { pos_part[b] = sp[0]; neg_part[b] = sn[0]; }
}

__global__ __launch_bounds__(256) void final_kernel(
    const float* __restrict__ pos_part, const float* __restrict__ neg_part,
    float* __restrict__ out) {
  __shared__ float sp[256], sn[256];
  const int t = threadIdx.x;
  float p = 0.0f, n = 0.0f;
  for (int i = t; i < NQU; i += 256) { p += pos_part[i]; n += neg_part[i]; }
  sp[t] = p; sn[t] = n;
  __syncthreads();
  for (int off = 128; off >= 1; off >>= 1) {
    if (t < off) { sp[t] += sp[t + off]; sn[t] += sn[t + off]; }
    __syncthreads();
  }
  if (t == 0) out[0] = log_sigmoid(sp[0]) + log_sigmoid(-sn[0]);
}

// ---------------------------------------------------------------------------
extern "C" void kernel_launch(void* const* d_in, const int* in_sizes, int n_in,
                              void* d_out, int out_size, void* d_ws, size_t ws_size,
                              hipStream_t stream) {
  (void)in_sizes; (void)n_in; (void)out_size; (void)ws_size;

  const int*   ru_idx = (const int*)  d_in[0];
  const int*   au_idx = (const int*)  d_in[1];
  const int*   rv_idx = (const int*)  d_in[2];
  const int*   av_idx = (const int*)  d_in[3];
  const int*   rn_idx = (const int*)  d_in[4];
  const int*   an_idx = (const int*)  d_in[5];
  const float* qu_seq = (const float*)d_in[6];
  const int*   qu_len = (const int*)  d_in[7];
  const float* qn_seq = (const float*)d_in[8];
  const int*   qn_len = (const int*)  d_in[9];
  const float* ru_emb = (const float*)d_in[10];
  const float* au_emb = (const float*)d_in[11];
  const float* rv_emb = (const float*)d_in[12];
  const float* av_emb = (const float*)d_in[13];
  const float* Wih_f  = (const float*)d_in[14];
  const float* Whh_f  = (const float*)d_in[15];
  const float* b_f    = (const float*)d_in[16];
  const float* Wih_b  = (const float*)d_in[17];
  const float* Whh_b  = (const float*)d_in[18];
  const float* b_b    = (const float*)d_in[19];
  const float* fc_W   = (const float*)d_in[20];
  const float* fc_b   = (const float*)d_in[21];

  char* ws = (char*)d_ws;
  size_t off = 0;
  auto take = [&](size_t bytes) -> void* {
    void* p = ws + off;
    off += (bytes + 255) & ~(size_t)255;
    return p;
  };
  unsigned short* wih_ws = (unsigned short*)take((size_t)2 * IPAD * G4 * 2);  // 655 KB
  unsigned short* whh_ws = (unsigned short*)take((size_t)2 * HDIM * G4 * 2);  // 262 KB
  unsigned short* fcw_ws = (unsigned short*)take((size_t)256 * 128 * 2);      //  65 KB
  unsigned short* hcat   = (unsigned short*)take((size_t)NSEQ * 256 * 2);     // 6.3 MB
  float*          emb    = (float*)         take((size_t)NSEQ * 128 * 4);     // 6.3 MB
  float*          pos_pt = (float*)         take((size_t)NQU * 4);
  float*          neg_pt = (float*)         take((size_t)NQU * 4);

  // Pre-swizzle weights into WMMA B-fragment layouts (bf16, K zero-padded).
  pack_w_bf16<<<40, 256, 0, stream>>>(Wih_f, wih_ws,              IDIM, IPAD, G4);
  pack_w_bf16<<<40, 256, 0, stream>>>(Wih_b, wih_ws + IPAD * G4,  IDIM, IPAD, G4);
  pack_w_bf16<<<16, 256, 0, stream>>>(Whh_f, whh_ws,              HDIM, HDIM, G4);
  pack_w_bf16<<<16, 256, 0, stream>>>(Whh_b, whh_ws + HDIM * G4,  HDIM, HDIM, G4);
  pack_w_bf16<<< 8, 256, 0, stream>>>(fc_W,  fcw_ws,              256,  256,  128);

  // Fused BiLSTM: (12288/16) tiles x 2 directions.
  lstm_kernel<<<(NSEQ / 16) * 2, 256, 0, stream>>>(qu_seq, qn_seq, qu_len, qn_len,
                                                   wih_ws, whh_ws, b_f, b_b, hcat);
  // FC to embeddings.
  fc_kernel<<<NSEQ / 16, 256, 0, stream>>>(hcat, fcw_ws, fc_b, emb);
  // Scoring + deterministic reductions.
  score_kernel<<<NQU, 128, 0, stream>>>(ru_idx, au_idx, rv_idx, av_idx, rn_idx, an_idx,
                                        ru_emb, au_emb, rv_emb, av_emb, emb, pos_pt, neg_pt);
  final_kernel<<<1, 256, 0, stream>>>(pos_pt, neg_pt, (float*)d_out);
}
